// AttentionNodeUpdateNet_74826920231621
// MI455X (gfx1250) — matrix operand order repaired and checked by reading, not compiled, not run on hardware
//
#include <hip/hip_runtime.h>
#include <hip/hip_bf16.h>

#define N_NODES 50000
#define N_EDGES 400000
#define D_      128
#define SPLIT_  32
#define EA_     64
#define IN_     192     // D + EA
#define H_      256
#define ROWSZ   160     // D + SPLIT
#define TM      16

typedef __bf16 bf16;
typedef __attribute__((ext_vector_type(16))) __bf16 v16bf;
typedef __attribute__((ext_vector_type(8)))  __bf16 v8bf;
typedef __attribute__((ext_vector_type(8)))  float  v8f;

// ---------- helpers ----------
__device__ __forceinline__ unsigned fkey(float x) {
    unsigned u = __float_as_uint(x);
    return (u & 0x80000000u) ? ~u : (u | 0x80000000u);
}
__device__ __forceinline__ float funkey(unsigned k) {
    unsigned u = (k & 0x80000000u) ? (k ^ 0x80000000u) : ~k;
    return __uint_as_float(u);
}
__device__ __forceinline__ v16bf cat8(v8bf lo, v8bf hi) {
    return __builtin_shufflevector(lo, hi, 0,1,2,3,4,5,6,7,8,9,10,11,12,13,14,15);
}
// A fragment (16x32 bf16) from LDS row-major [STRIDE]; layout per CDNA5 ISA:
// lanes 0-15: row=lane, elems 0-7 -> K k0+0..7, elems 8-15 -> K k0+16..23
// lanes16-31: row=lane-16, elems 0-7 -> K k0+8..15, elems 8-15 -> K k0+24..31
template<int STRIDE>
__device__ __forceinline__ v16bf loadA(const bf16* A, int lane, int k0) {
    int r  = lane & 15;
    int hh = (lane >> 4) << 3;   // 0 or 8
    const bf16* p = A + r * STRIDE + k0;
    return cat8(*(const v8bf*)(p + hh), *(const v8bf*)(p + 16 + hh));
}
// B fragment from pre-swizzled weights: contiguous 16 bf16 per lane
__device__ __forceinline__ v16bf loadB(const bf16* Ws, int ks, int ntiles, int nt, int lane) {
    const bf16* p = Ws + ((size_t)((ks * ntiles + nt) * 32 + lane) << 4);
    return cat8(*(const v8bf*)p, *(const v8bf*)(p + 8));
}
__device__ __forceinline__ v8f wmma_bf16(v16bf a, v16bf b, v8f c) {
    return __builtin_amdgcn_wmma_f32_16x16x32_bf16(false, a, false, b, (short)0, c, false, false);
}

// ---------- prep kernels ----------
__global__ __launch_bounds__(256) void k_zero(float* flow, unsigned* mkey, float* sume) {
    int i = blockIdx.x * 256 + threadIdx.x;
    if (i < N_NODES * 2 * D_) flow[i] = 0.f;
    if (i < 2) { mkey[i] = 0u; sume[i] = 0.f; }
}

// fp32 (K x N) row-major -> bf16 swizzled [ks][nt][lane][16] in WMMA B layout:
// elems 0-7: K = ks*32 + 8*(lane>>4) + elem ; elems 8-15: K = ks*32 + 16 + 8*(lane>>4) + (elem-8) ; col = nt*16 + (lane&15)
__global__ __launch_bounds__(256) void k_swizzle(const float* __restrict__ w,
                                                 bf16* __restrict__ ws, int K, int N) {
    int idx = blockIdx.x * 256 + threadIdx.x;
    if (idx >= K * N) return;
    int e    = idx & 15;
    int lane = (idx >> 4) & 31;
    int t    = idx >> 9;
    int ntiles = N >> 4;
    int nt = t % ntiles;
    int ks = t / ntiles;
    int kk = ks * 32 + ((lane >> 4) << 3) + (e & 7) + ((e >> 3) << 4);
    int cc = nt * 16 + (lane & 15);
    ws[idx] = (bf16)w[(size_t)kk * N + cc];
}

// per-node partial dots with w_att (rank-1 attention)
__global__ __launch_bounds__(256) void k_scores(const float* __restrict__ x,
                                                const float* __restrict__ watt,
                                                float* __restrict__ scol,
                                                float* __restrict__ srow) {
    int node = blockIdx.x * 8 + (threadIdx.x >> 5);
    int lane = threadIdx.x & 31;
    if (node >= N_NODES) return;
    const float* xp = x + (size_t)node * ROWSZ;
    float a = 0.f, b = 0.f;
    #pragma unroll
    for (int i = lane; i < D_; i += 32) {
        float xv = xp[i];
        a += xv * watt[i];
        b += xv * watt[D_ + i];
    }
    #pragma unroll
    for (int m = 1; m < 32; m <<= 1) { a += __shfl_xor(a, m, 32); b += __shfl_xor(b, m, 32); }
    if (lane == 0) { scol[node] = a; srow[node] = b; }
}

// logits + global masked max per direction (dir0: row>col t2d, dir1: row<col d2t)
__global__ __launch_bounds__(256) void k_logits(const int* __restrict__ row,
                                                const int* __restrict__ col,
                                                const float* __restrict__ scol,
                                                const float* __restrict__ srow,
                                                const float* __restrict__ batt,
                                                float* __restrict__ logits,
                                                unsigned* __restrict__ mkey) {
    __shared__ unsigned bmax[2];
    if (threadIdx.x < 2) bmax[threadIdx.x] = 0u;
    __syncthreads();
    int e = blockIdx.x * 256 + threadIdx.x;
    unsigned k0 = 0u, k1 = 0u;
    if (e < N_EDGES) {
        int r = row[e], c = col[e];
        float z  = scol[c] + srow[r] + batt[0];
        float lg = (z >= 0.f) ? z : 0.01f * z;   // leaky_relu
        logits[e] = lg;
        unsigned kk = fkey(lg);
        if (r > c) k0 = kk;
        if (r < c) k1 = kk;
    }
    #pragma unroll
    for (int m = 1; m < 32; m <<= 1) {
        k0 = max(k0, __shfl_xor(k0, m, 32));
        k1 = max(k1, __shfl_xor(k1, m, 32));
    }
    if ((threadIdx.x & 31) == 0) { atomicMax(&bmax[0], k0); atomicMax(&bmax[1], k1); }
    __syncthreads();
    if (threadIdx.x < 2) atomicMax(&mkey[threadIdx.x], bmax[threadIdx.x]);
}

__global__ __launch_bounds__(256) void k_expsum(const int* __restrict__ row,
                                                const int* __restrict__ col,
                                                const float* __restrict__ logits,
                                                const unsigned* __restrict__ mkey,
                                                float* __restrict__ sume) {
    int e = blockIdx.x * 256 + threadIdx.x;
    float s0 = 0.f, s1 = 0.f;
    if (e < N_EDGES) {
        int r = row[e], c = col[e];
        float lg = logits[e];
        if (r > c) s0 = expf(lg - funkey(mkey[0]));
        if (r < c) s1 = expf(lg - funkey(mkey[1]));
    }
    #pragma unroll
    for (int m = 1; m < 32; m <<= 1) { s0 += __shfl_xor(s0, m, 32); s1 += __shfl_xor(s1, m, 32); }
    if ((threadIdx.x & 31) == 0) {
        if (s0 != 0.f) atomicAdd(&sume[0], s0);
        if (s1 != 0.f) atomicAdd(&sume[1], s1);
    }
}

// ---------- edge MLP: 16 edges/tile, 4 waves, bf16 WMMA ----------
struct EdgeP {
    const float* x; const int* row; const int* col; const float* ea;
    const float* logits; const unsigned* mkey; const float* sume;
    const bf16* W1s[2]; const float* b1[2]; const float* g1[2]; const float* be1[2];
    const bf16* W2s[2]; const float* b2[2]; const float* g2[2]; const float* be2[2];
    float* flow;
};

__global__ __launch_bounds__(128) void k_edge_mlp(EdgeP P) {
    const int dir  = blockIdx.y;          // 0=t2d(row>col)->flow[0:128) ; 1=d2t(row<col)->flow[128:256)
    const int e0   = blockIdx.x * TM;
    const int tid  = threadIdx.x;
    const int wave = tid >> 5, lane = tid & 31;

    __shared__ bf16  A1[TM * 200];        // 16 x 192 (pad 200)
    __shared__ float C1[TM * H_];         // 16 x 256
    __shared__ bf16  A2[TM * 264];        // 16 x 256 (pad 264)
    __shared__ float C2[TM * D_];         // 16 x 128
    __shared__ float attn_s[TM];
    __shared__ int   node_s[TM], cnode_s[TM], valid_s[TM];

    if (tid < TM) {
        int e = e0 + tid;
        int r = P.row[e], c = P.col[e];
        bool m = dir ? (r < c) : (r > c);
        float a = 0.f;
        if (m) a = expf(P.logits[e] - funkey(P.mkey[dir])) / P.sume[dir];
        attn_s[tid] = a; node_s[tid] = r; cnode_s[tid] = c; valid_s[tid] = m ? 1 : 0;
    }
    __syncthreads();

    // A1 = attn * [xn[col], edge_attr]   (fp32 gather -> bf16)
    for (int idx = tid; idx < TM * IN_; idx += 128) {
        int r = idx / IN_, cc = idx - r * IN_;
        float v = (cc < D_) ? P.x[(size_t)cnode_s[r] * ROWSZ + cc]
                            : P.ea[(size_t)(e0 + r) * EA_ + (cc - D_)];
        A1[r * 200 + cc] = (bf16)(attn_s[r] * v);
    }
    __syncthreads();

    // GEMM1: (16x192) @ (192x256); each wave owns 4 N-tiles of 16
    v8f acc[4] = {};
    #pragma unroll
    for (int ks = 0; ks < 6; ++ks) {
        v16bf a = loadA<200>(A1, lane, ks * 32);
        #pragma unroll
        for (int j = 0; j < 4; ++j) {
            v16bf b = loadB(P.W1s[dir], ks, 16, wave * 4 + j, lane);
            acc[j] = wmma_bf16(a, b, acc[j]);
        }
    }
    {
        const float* b1 = P.b1[dir];
        int rb = (lane < 16) ? 0 : 8;
        #pragma unroll
        for (int j = 0; j < 4; ++j) {
            int cc = (wave * 4 + j) * 16 + (lane & 15);
            #pragma unroll
            for (int v = 0; v < 8; ++v) C1[(rb + v) * H_ + cc] = acc[j][v] + b1[cc];
        }
    }
    __syncthreads();

    // LN1 + relu -> A2 (bf16). 8 threads per row, shfl reduce inside 8-lane groups.
    {
        int r = tid >> 3, p = tid & 7;
        float s = 0.f, s2 = 0.f;
        #pragma unroll
        for (int i = 0; i < 32; ++i) { float xv = C1[r * H_ + p * 32 + i]; s += xv; s2 += xv * xv; }
        #pragma unroll
        for (int m = 1; m < 8; m <<= 1) { s += __shfl_xor(s, m, 32); s2 += __shfl_xor(s2, m, 32); }
        float mean = s * (1.f / H_);
        float var  = s2 * (1.f / H_) - mean * mean;
        float rstd = rsqrtf(var + 1e-5f);
        const float* g = P.g1[dir]; const float* be = P.be1[dir];
        #pragma unroll
        for (int i = 0; i < 32; ++i) {
            int cc = p * 32 + i;
            float xv = (C1[r * H_ + cc] - mean) * rstd * g[cc] + be[cc];
            A2[r * 264 + cc] = (bf16)fmaxf(xv, 0.f);
        }
    }
    __syncthreads();

    // GEMM2: (16x256) @ (256x128); each wave owns 2 N-tiles
    v8f acc2[2] = {};
    #pragma unroll
    for (int ks = 0; ks < 8; ++ks) {
        v16bf a = loadA<264>(A2, lane, ks * 32);
        #pragma unroll
        for (int j = 0; j < 2; ++j) {
            v16bf b = loadB(P.W2s[dir], ks, 8, wave * 2 + j, lane);
            acc2[j] = wmma_bf16(a, b, acc2[j]);
        }
    }
    {
        const float* b2 = P.b2[dir];
        int rb = (lane < 16) ? 0 : 8;
        #pragma unroll
        for (int j = 0; j < 2; ++j) {
            int cc = (wave * 2 + j) * 16 + (lane & 15);
            #pragma unroll
            for (int v = 0; v < 8; ++v) C2[(rb + v) * D_ + cc] = acc2[j][v] + b2[cc];
        }
    }
    __syncthreads();

    // LN2 + relu + masked atomic scatter (segment_sum over row)
    {
        int r = tid >> 3, p = tid & 7;
        float s = 0.f, s2 = 0.f;
        #pragma unroll
        for (int i = 0; i < 16; ++i) { float xv = C2[r * D_ + p * 16 + i]; s += xv; s2 += xv * xv; }
        #pragma unroll
        for (int m = 1; m < 8; m <<= 1) { s += __shfl_xor(s, m, 32); s2 += __shfl_xor(s2, m, 32); }
        float mean = s * (1.f / D_);
        float var  = s2 * (1.f / D_) - mean * mean;
        float rstd = rsqrtf(var + 1e-5f);
        if (valid_s[r]) {
            const float* g = P.g2[dir]; const float* be = P.be2[dir];
            float* dst = P.flow + (size_t)node_s[r] * (2 * D_) + (dir ? D_ : 0);
            #pragma unroll
            for (int i = 0; i < 16; ++i) {
                int cc = p * 16 + i;
                float xv = (C2[r * D_ + cc] - mean) * rstd * g[cc] + be[cc];
                atomicAdd(dst + cc, fmaxf(xv, 0.f));
            }
        }
    }
}

// ---------- node update: flow-block + self-block ----------
struct NodeP {
    const float* x; const float* flow;
    const bf16* NW; const float* nb; const float* ng; const float* nbe;
    const bf16* SW; const float* sb; const float* sg; const float* sbe;
    float* out;
};

__global__ __launch_bounds__(128) void k_node_update(NodeP P) {
    const int n0  = blockIdx.x * TM;
    const int tid = threadIdx.x, wave = tid >> 5, lane = tid & 31;
    __shared__ bf16  Af[TM * 264];   // flow tile 16x256
    __shared__ float C[TM * D_];
    __shared__ float U[TM * D_];
    __shared__ bf16  Ax[TM * 136];   // xn tile 16x128

    for (int idx = tid; idx < TM * H_; idx += 128) {
        int r = idx >> 8, cc = idx & 255;
        Af[r * 264 + cc] = (bf16)P.flow[(size_t)(n0 + r) * H_ + cc];
    }
    for (int idx = tid; idx < TM * D_; idx += 128) {
        int r = idx >> 7, cc = idx & 127;
        Ax[r * 136 + cc] = (bf16)P.x[(size_t)(n0 + r) * ROWSZ + cc];
    }
    __syncthreads();

    // node block: (16x256)@(256x128)
    v8f acc[2] = {};
    #pragma unroll
    for (int ks = 0; ks < 8; ++ks) {
        v16bf a = loadA<264>(Af, lane, ks * 32);
        #pragma unroll
        for (int j = 0; j < 2; ++j) {
            v16bf b = loadB(P.NW, ks, 8, wave * 2 + j, lane);
            acc[j] = wmma_bf16(a, b, acc[j]);
        }
    }
    {
        int rb = (lane < 16) ? 0 : 8;
        #pragma unroll
        for (int j = 0; j < 2; ++j) {
            int cc = (wave * 2 + j) * 16 + (lane & 15);
            #pragma unroll
            for (int v = 0; v < 8; ++v) C[(rb + v) * D_ + cc] = acc[j][v] + P.nb[cc];
        }
    }
    __syncthreads();
    {
        int r = tid >> 3, p = tid & 7;
        float s = 0.f, s2 = 0.f;
        #pragma unroll
        for (int i = 0; i < 16; ++i) { float xv = C[r * D_ + p * 16 + i]; s += xv; s2 += xv * xv; }
        #pragma unroll
        for (int m = 1; m < 8; m <<= 1) { s += __shfl_xor(s, m, 32); s2 += __shfl_xor(s2, m, 32); }
        float mean = s * (1.f / D_), var = s2 * (1.f / D_) - mean * mean;
        float rstd = rsqrtf(var + 1e-5f);
        #pragma unroll
        for (int i = 0; i < 16; ++i) {
            int cc = p * 16 + i;
            U[r * D_ + cc] = fmaxf((C[r * D_ + cc] - mean) * rstd * P.ng[cc] + P.nbe[cc], 0.f);
        }
    }
    __syncthreads();

    // self block: (16x128)@(128x128)
    v8f acc2[2] = {};
    #pragma unroll
    for (int ks = 0; ks < 4; ++ks) {
        v16bf a = loadA<136>(Ax, lane, ks * 32);
        #pragma unroll
        for (int j = 0; j < 2; ++j) {
            v16bf b = loadB(P.SW, ks, 8, wave * 2 + j, lane);
            acc2[j] = wmma_bf16(a, b, acc2[j]);
        }
    }
    {
        int rb = (lane < 16) ? 0 : 8;
        #pragma unroll
        for (int j = 0; j < 2; ++j) {
            int cc = (wave * 2 + j) * 16 + (lane & 15);
            #pragma unroll
            for (int v = 0; v < 8; ++v) C[(rb + v) * D_ + cc] = acc2[j][v] + P.sb[cc];
        }
    }
    __syncthreads();
    {
        int r = tid >> 3, p = tid & 7;
        float s = 0.f, s2 = 0.f;
        #pragma unroll
        for (int i = 0; i < 16; ++i) { float xv = C[r * D_ + p * 16 + i]; s += xv; s2 += xv * xv; }
        #pragma unroll
        for (int m = 1; m < 8; m <<= 1) { s += __shfl_xor(s, m, 32); s2 += __shfl_xor(s2, m, 32); }
        float mean = s * (1.f / D_), var = s2 * (1.f / D_) - mean * mean;
        float rstd = rsqrtf(var + 1e-5f);
        size_t node = (size_t)(n0 + r);
        #pragma unroll
        for (int i = 0; i < 16; ++i) {
            int cc = p * 16 + i;
            float sv = fmaxf((C[r * D_ + cc] - mean) * rstd * P.sg[cc] + P.sbe[cc], 0.f);
            P.out[node * ROWSZ + cc] = U[r * D_ + cc] + sv;
        }
    }
    // passthrough split features
    for (int idx = tid; idx < TM * SPLIT_; idx += 128) {
        int r = idx >> 5, cc = idx & 31;
        P.out[(size_t)(n0 + r) * ROWSZ + D_ + cc] = P.x[(size_t)(n0 + r) * ROWSZ + D_ + cc];
    }
}

// ---------- host launcher ----------
extern "C" void kernel_launch(void* const* d_in, const int* in_sizes, int n_in,
                              void* d_out, int out_size, void* d_ws, size_t ws_size,
                              hipStream_t stream) {
    (void)in_sizes; (void)n_in; (void)out_size; (void)ws_size;
    const float* x    = (const float*)d_in[0];
    const int*   ei   = (const int*)d_in[1];
    const float* ea   = (const float*)d_in[2];
    const float* watt = (const float*)d_in[3];
    const float* batt = (const float*)d_in[4];
    const int* row = ei;
    const int* col = ei + N_EDGES;

    char* wsb = (char*)d_ws;
    size_t off = 0;
    auto alloc = [&](size_t bytes) -> char* {
        char* p = wsb + off;
        off = (off + bytes + 255) & ~(size_t)255;
        return p;
    };
    float*    flow   = (float*)alloc((size_t)N_NODES * 2 * D_ * 4);
    float*    logits = (float*)alloc((size_t)N_EDGES * 4);
    float*    scol   = (float*)alloc((size_t)N_NODES * 4);
    float*    srow   = (float*)alloc((size_t)N_NODES * 4);
    unsigned* mkey   = (unsigned*)alloc(2 * 4);
    float*    sume   = (float*)alloc(2 * 4);
    bf16* W1s0 = (bf16*)alloc((size_t)IN_ * H_ * 2);   // t2d
    bf16* W1s1 = (bf16*)alloc((size_t)IN_ * H_ * 2);   // d2t
    bf16* W2s0 = (bf16*)alloc((size_t)H_ * D_ * 2);
    bf16* W2s1 = (bf16*)alloc((size_t)H_ * D_ * 2);
    bf16* NWs  = (bf16*)alloc((size_t)H_ * D_ * 2);
    bf16* SWs  = (bf16*)alloc((size_t)D_ * D_ * 2);

    k_zero<<<(N_NODES * 2 * D_ + 255) / 256, 256, 0, stream>>>(flow, mkey, sume);

    k_swizzle<<<(IN_ * H_) / 256, 256, 0, stream>>>((const float*)d_in[13], W1s0, IN_, H_); // t2d_w1
    k_swizzle<<<(IN_ * H_) / 256, 256, 0, stream>>>((const float*)d_in[5],  W1s1, IN_, H_); // d2t_w1
    k_swizzle<<<(H_ * D_)  / 256, 256, 0, stream>>>((const float*)d_in[17], W2s0, H_, D_);  // t2d_w2
    k_swizzle<<<(H_ * D_)  / 256, 256, 0, stream>>>((const float*)d_in[9],  W2s1, H_, D_);  // d2t_w2
    k_swizzle<<<(H_ * D_)  / 256, 256, 0, stream>>>((const float*)d_in[21], NWs,  H_, D_);  // node_w
    k_swizzle<<<(D_ * D_)  / 256, 256, 0, stream>>>((const float*)d_in[25], SWs,  D_, D_);  // self_w

    k_scores<<<N_NODES / 8, 256, 0, stream>>>(x, watt, scol, srow);
    k_logits<<<(N_EDGES + 255) / 256, 256, 0, stream>>>(row, col, scol, srow, batt, logits, mkey);
    k_expsum<<<(N_EDGES + 255) / 256, 256, 0, stream>>>(row, col, logits, mkey, sume);

    EdgeP ep;
    ep.x = x; ep.row = row; ep.col = col; ep.ea = ea;
    ep.logits = logits; ep.mkey = mkey; ep.sume = sume;
    ep.W1s[0] = W1s0;                      ep.W1s[1] = W1s1;
    ep.b1[0]  = (const float*)d_in[14];    ep.b1[1]  = (const float*)d_in[6];
    ep.g1[0]  = (const float*)d_in[15];    ep.g1[1]  = (const float*)d_in[7];
    ep.be1[0] = (const float*)d_in[16];    ep.be1[1] = (const float*)d_in[8];
    ep.W2s[0] = W2s0;                      ep.W2s[1] = W2s1;
    ep.b2[0]  = (const float*)d_in[18];    ep.b2[1]  = (const float*)d_in[10];
    ep.g2[0]  = (const float*)d_in[19];    ep.g2[1]  = (const float*)d_in[11];
    ep.be2[0] = (const float*)d_in[20];    ep.be2[1] = (const float*)d_in[12];
    ep.flow = flow;
    dim3 eg(N_EDGES / TM, 2);
    k_edge_mlp<<<eg, 128, 0, stream>>>(ep);

    NodeP np;
    np.x = x; np.flow = flow;
    np.NW = NWs; np.nb = (const float*)d_in[22]; np.ng = (const float*)d_in[23]; np.nbe = (const float*)d_in[24];
    np.SW = SWs; np.sb = (const float*)d_in[26]; np.sg = (const float*)d_in[27]; np.sbe = (const float*)d_in[28];
    np.out = (float*)d_out;
    k_node_update<<<N_NODES / TM, 128, 0, stream>>>(np);
}
